// DecoderOnlyAttention_86560770883870
// MI455X (gfx1250) — matrix operand order repaired and checked by reading, not compiled, not run on hardware
//
#include <hip/hip_runtime.h>
#include <hip/hip_bf16.h>

typedef __attribute__((ext_vector_type(16))) _Float16 v16h;
typedef __attribute__((ext_vector_type(8)))  float    v8f;

#define SEQ   2048
#define DM    1024
#define NH    16
#define DH    64
#define IGNORE_VAL -100000.0f

// A fragment: 16x32 f16, lane row = lane&15, contiguous 16 K halves (k = 16*g + t).
__device__ __forceinline__ v16h load_afrag(const _Float16* __restrict__ row,
                                           int kb, int g)
{
    v16h a;
    #pragma unroll
    for (int t = 0; t < 16; t++) a[t] = row[kb + 16*g + t];
    return a;
}

// B fragment: 32x16 f16, lane col pre-applied in `col`, contiguous 16 K halves.
__device__ __forceinline__ v16h load_bfrag(const _Float16* __restrict__ col,
                                           int kb)
{
    v16h b;
    #pragma unroll
    for (int t = 0; t < 16; t++) b[t] = col[kb + t];
    return b;
}

// ---------------------------------------------------------------------------
// Kernel 0: preprocessing.
//  - x   [2][2048][1024] f32  -> xh  f16 (same layout)
//  - Qs/Ks/Vs [h][d][k] f32   -> wq/wk/wv [h][k][d] f16   (transposed)
//  - O   [hk][d] f32          -> ot [d][hk] f16           (transposed)
// ---------------------------------------------------------------------------
__global__ __launch_bounds__(256) void prep_kernel(
    const float* __restrict__ x,
    const float* __restrict__ Qs, const float* __restrict__ Ks,
    const float* __restrict__ Vs, const float* __restrict__ O,
    _Float16* __restrict__ xh, _Float16* __restrict__ wq,
    _Float16* __restrict__ wk, _Float16* __restrict__ wv,
    _Float16* __restrict__ ot)
{
    size_t i = (size_t)blockIdx.x * 256 + threadIdx.x;
    const size_t MX = (size_t)2 * SEQ * DM;   // 4M
    const size_t MW = (size_t)NH * DM * DH;   // 1M

    if (i < MX) { xh[i] = (_Float16)x[i]; return; }
    i -= MX;
    if (i < MW) {           // wq[i], i = (h*64+k)*1024 + d
        const size_t d = i & 1023, hk = i >> 10, h = hk >> 6, k = hk & 63;
        wq[i] = (_Float16)Qs[(h * DM + d) * DH + k]; return;
    }
    i -= MW;
    if (i < MW) {
        const size_t d = i & 1023, hk = i >> 10, h = hk >> 6, k = hk & 63;
        wk[i] = (_Float16)Ks[(h * DM + d) * DH + k]; return;
    }
    i -= MW;
    if (i < MW) {
        const size_t d = i & 1023, hk = i >> 10, h = hk >> 6, k = hk & 63;
        wv[i] = (_Float16)Vs[(h * DM + d) * DH + k]; return;
    }
    i -= MW;
    if (i < MW) {           // ot[i], i = d*1024 + hk
        const size_t hk = i & 1023, d = i >> 10;
        ot[i] = (_Float16)O[hk * DM + d]; return;
    }
}

// ---------------------------------------------------------------------------
// Kernel 1: fused Q/K/V projection.  One wave -> one 16x64 head tile,
// software-pipelined K loop (double-buffered fragments).
// q,k stored f16 [b][p][h][64];  v stored TRANSPOSED f16 [b][h][64][seq].
// ---------------------------------------------------------------------------
__global__ __launch_bounds__(128) void qkv_proj_kernel(
    const _Float16* __restrict__ xh,
    const _Float16* __restrict__ wq, const _Float16* __restrict__ wk,
    const _Float16* __restrict__ wv,
    const float* __restrict__ Qbs, const float* __restrict__ Kbs,
    const float* __restrict__ Vbs,
    _Float16* __restrict__ qbuf, _Float16* __restrict__ kbuf,
    _Float16* __restrict__ vtb)
{
    const int wid  = blockIdx.x * 4 + (threadIdx.x >> 5);
    const int lane = threadIdx.x & 31;
    const int head = wid & 15;
    const int pt   = (wid >> 4) & 127;
    const int b    = (wid >> 11) & 1;
    const int mat  = wid >> 12;           // 0=Q 1=K 2=V

    const _Float16* W    = (mat == 0) ? wq  : (mat == 1) ? wk  : wv;
    const float*    bias = (mat == 0) ? Qbs : (mat == 1) ? Kbs : Vbs;

    const int p0 = pt * 16, nn = lane & 15, g = lane >> 4;

    v8f acc[4];
    #pragma unroll
    for (int kt = 0; kt < 4; kt++) acc[kt] = (v8f){0.f,0.f,0.f,0.f,0.f,0.f,0.f,0.f};

    const _Float16* xr = xh + ((size_t)b * SEQ + p0 + nn) * DM;   // A row m=nn
    const _Float16* wt = W + (size_t)head * DH * DM;              // [k][d]
    const _Float16* wcol[4];
    #pragma unroll
    for (int kt = 0; kt < 4; kt++) wcol[kt] = wt + (size_t)(kt*16 + nn) * DM + 16*g;

    // ---- software-pipelined K loop: stage0/stage1 double buffering ----
    v16h a0, a1, b0[4], b1[4];
    a0 = load_afrag(xr, 0, g);
    #pragma unroll
    for (int kt = 0; kt < 4; kt++) b0[kt] = load_bfrag(wcol[kt], 0);

    for (int kb = 0; kb < DM - 64; kb += 64) {
        a1 = load_afrag(xr, kb + 32, g);
        #pragma unroll
        for (int kt = 0; kt < 4; kt++) b1[kt] = load_bfrag(wcol[kt], kb + 32);
        #pragma unroll
        for (int kt = 0; kt < 4; kt++)
            acc[kt] = __builtin_amdgcn_wmma_f32_16x16x32_f16(false, a0, false, b0[kt],
                                                             (short)0, acc[kt], false, false);
        a0 = load_afrag(xr, kb + 64, g);
        #pragma unroll
        for (int kt = 0; kt < 4; kt++) b0[kt] = load_bfrag(wcol[kt], kb + 64);
        #pragma unroll
        for (int kt = 0; kt < 4; kt++)
            acc[kt] = __builtin_amdgcn_wmma_f32_16x16x32_f16(false, a1, false, b1[kt],
                                                             (short)0, acc[kt], false, false);
    }
    // tail: kb = DM-64 and DM-32
    a1 = load_afrag(xr, DM - 32, g);
    #pragma unroll
    for (int kt = 0; kt < 4; kt++) b1[kt] = load_bfrag(wcol[kt], DM - 32);
    #pragma unroll
    for (int kt = 0; kt < 4; kt++)
        acc[kt] = __builtin_amdgcn_wmma_f32_16x16x32_f16(false, a0, false, b0[kt],
                                                         (short)0, acc[kt], false, false);
    #pragma unroll
    for (int kt = 0; kt < 4; kt++)
        acc[kt] = __builtin_amdgcn_wmma_f32_16x16x32_f16(false, a1, false, b1[kt],
                                                         (short)0, acc[kt], false, false);

    const float scale = (mat == 0) ? 0.125f : 1.0f;   // 1/sqrt(64) folded into q
    #pragma unroll
    for (int kt = 0; kt < 4; kt++) {
        const float bv = bias[head * DH + kt*16 + nn];
        #pragma unroll
        for (int i = 0; i < 8; i++) {
            const float val = (acc[kt][i] + bv) * scale;
            const int   p   = p0 + i + 8*g;
            if (mat == 0)
                qbuf[(((size_t)b * SEQ + p) * NH + head) * DH + kt*16 + nn] = (_Float16)val;
            else if (mat == 1)
                kbuf[(((size_t)b * SEQ + p) * NH + head) * DH + kt*16 + nn] = (_Float16)val;
            else
                vtb[(((size_t)b * NH + head) * DH + kt*16 + nn) * SEQ + p]  = (_Float16)val;
        }
    }
}

// ---------------------------------------------------------------------------
// Kernel 2: causal flash attention.  One wave per (b, head, 16-row q tile).
// Online softmax over 32-wide key blocks; LDS transposes exp(S) into the
// WMMA A layout.  V is read from the transposed buffer (contiguous-in-K).
// ---------------------------------------------------------------------------
__global__ __launch_bounds__(32) void attn_kernel(
    const _Float16* __restrict__ qbuf, const _Float16* __restrict__ kbuf,
    const _Float16* __restrict__ vtb, _Float16* __restrict__ zbuf)
{
    __shared__ _Float16 sP[16 * 32];          // one wave per workgroup

    const int wid  = blockIdx.x;              // 0..4095
    const int qt   = wid & 127;
    const int head = (wid >> 7) & 15;
    const int b    = wid >> 11;
    const int lane = threadIdx.x & 31;
    const int nn   = lane & 15, g = lane >> 4;
    const int p0   = qt * 16;

    // Q A-fragments (two K=32 chunks of the 64-wide head dim), pre-scaled.
    v16h aq[2];
    const _Float16* qrow = qbuf + (((size_t)b * SEQ + p0 + nn) * NH + head) * DH;
    #pragma unroll
    for (int c = 0; c < 2; c++) aq[c] = load_afrag(qrow, 32*c, g);

    const _Float16* vbase = vtb + ((size_t)b * NH + head) * DH * SEQ;

    float m_st[8], l_st[8];
    v8f   o_acc[4];
    #pragma unroll
    for (int i = 0; i < 8; i++) { m_st[i] = -3.0e38f; l_st[i] = 0.f; }
    #pragma unroll
    for (int vt = 0; vt < 4; vt++) o_acc[vt] = (v8f){0.f,0.f,0.f,0.f,0.f,0.f,0.f,0.f};

    const int nblocks = (p0 + 15) / 32 + 1;   // causal: only s <= p0+15 needed
    for (int sb = 0; sb < nblocks; sb++) {
        const int s0 = sb * 32;

        // prefetch next key block into cache (global_prefetch_b8)
        if (sb + 1 < nblocks) {
            const _Float16* nk =
                kbuf + (((size_t)b * SEQ + s0 + 32 + nn) * NH + head) * DH;
            __builtin_prefetch(nk, 0, 3);
        }

        // ---- scores: two 16x16 tiles (j = s sub-tile), K-dim = head dim ----
        v8f sc[2];
        #pragma unroll
        for (int j = 0; j < 2; j++) {
            sc[j] = (v8f){0.f,0.f,0.f,0.f,0.f,0.f,0.f,0.f};
            const _Float16* krow =
                kbuf + (((size_t)b * SEQ + s0 + 16*j + nn) * NH + head) * DH + 16*g;
            #pragma unroll
            for (int c = 0; c < 2; c++) {
                const v16h bk = load_bfrag(krow, 32*c);
                sc[j] = __builtin_amdgcn_wmma_f32_16x16x32_f16(false, aq[c], false, bk,
                                                               (short)0, sc[j], false, false);
            }
        }

        // ---- causal mask (C layout: col = nn, row = i + 8g) ----
        #pragma unroll
        for (int j = 0; j < 2; j++)
            #pragma unroll
            for (int i = 0; i < 8; i++)
                if (s0 + 16*j + nn > p0 + i + 8*g) sc[j][i] = IGNORE_VAL;

        // ---- online softmax: row max across 16 lanes ----
        float mnew[8], corr[8], psum[8];
        #pragma unroll
        for (int i = 0; i < 8; i++) {
            float v = fmaxf(sc[0][i], sc[1][i]);
            #pragma unroll
            for (int m = 1; m < 16; m <<= 1) v = fmaxf(v, __shfl_xor(v, m, 32));
            mnew[i] = fmaxf(m_st[i], v);
            corr[i] = __expf(m_st[i] - mnew[i]);
        }

        // ---- P = exp(S - m); write to LDS in (row, s) order ----
        #pragma unroll
        for (int i = 0; i < 8; i++) {
            const float e0 = __expf(sc[0][i] - mnew[i]);
            const float e1 = __expf(sc[1][i] - mnew[i]);
            sP[(i + 8*g) * 32 + nn]      = (_Float16)e0;
            sP[(i + 8*g) * 32 + 16 + nn] = (_Float16)e1;
            psum[i] = e0 + e1;
        }

        // ---- row sums + state update + rescale accumulators ----
        #pragma unroll
        for (int i = 0; i < 8; i++) {
            float s = psum[i];
            #pragma unroll
            for (int m = 1; m < 16; m <<= 1) s += __shfl_xor(s, m, 32);
            l_st[i] = l_st[i] * corr[i] + s;
            m_st[i] = mnew[i];
        }
        #pragma unroll
        for (int vt = 0; vt < 4; vt++)
            #pragma unroll
            for (int i = 0; i < 8; i++) o_acc[vt][i] *= corr[i];

        // ---- LDS fence, read P back as a 16x32 A fragment ----
        asm volatile("s_wait_dscnt 0" ::: "memory");
        const v16h ap = load_afrag(sP + nn * 32, 0, g);

        // ---- P (16x32) @ V (32x64): 4 WMMAs, V^T gives contiguous K ----
        #pragma unroll
        for (int vt = 0; vt < 4; vt++) {
            const _Float16* vr = vbase + (size_t)(vt*16 + nn) * SEQ + 16*g;
            const v16h bv = load_bfrag(vr, s0);
            o_acc[vt] = __builtin_amdgcn_wmma_f32_16x16x32_f16(false, ap, false, bv,
                                                               (short)0, o_acc[vt], false, false);
        }
    }

    // ---- normalize and store z (f16, [b][p][h][64]) ----
    #pragma unroll
    for (int i = 0; i < 8; i++) {
        const float inv = 1.0f / l_st[i];
        _Float16* zr = zbuf + (((size_t)b * SEQ + p0 + i + 8*g) * NH + head) * DH;
        #pragma unroll
        for (int vt = 0; vt < 4; vt++)
            zr[vt*16 + nn] = (_Float16)(o_acc[vt][i] * inv);
    }
}

// ---------------------------------------------------------------------------
// Kernel 3: output projection.  z [4096 x 1024] (f16) @ O^T-layout + Ob.
// One wave -> 16x64 output tile, software-pipelined K loop.
// ---------------------------------------------------------------------------
__global__ __launch_bounds__(128) void out_proj_kernel(
    const _Float16* __restrict__ zbuf, const _Float16* __restrict__ ot,
    const float* __restrict__ Ob, float* __restrict__ out)
{
    const int wid  = blockIdx.x * 4 + (threadIdx.x >> 5);
    const int lane = threadIdx.x & 31;
    const int nn   = lane & 15, g = lane >> 4;
    const int dq   = wid & 15;            // 16 tiles of 64 columns
    const int rt   = (wid >> 4) & 255;    // 256 row tiles (b*p rows)
    const int d0 = dq * 64, r0 = rt * 16;

    v8f acc[4];
    #pragma unroll
    for (int j = 0; j < 4; j++) acc[j] = (v8f){0.f,0.f,0.f,0.f,0.f,0.f,0.f,0.f};

    const _Float16* zr = zbuf + (size_t)(r0 + nn) * DM;   // A row m = nn
    const _Float16* ocol[4];
    #pragma unroll
    for (int j = 0; j < 4; j++) ocol[j] = ot + (size_t)(d0 + j*16 + nn) * DM + 16*g;

    v16h a0, a1, b0[4], b1[4];
    a0 = load_afrag(zr, 0, g);
    #pragma unroll
    for (int j = 0; j < 4; j++) b0[j] = load_bfrag(ocol[j], 0);

    for (int kb = 0; kb < DM - 64; kb += 64) {
        a1 = load_afrag(zr, kb + 32, g);
        #pragma unroll
        for (int j = 0; j < 4; j++) b1[j] = load_bfrag(ocol[j], kb + 32);
        #pragma unroll
        for (int j = 0; j < 4; j++)
            acc[j] = __builtin_amdgcn_wmma_f32_16x16x32_f16(false, a0, false, b0[j],
                                                            (short)0, acc[j], false, false);
        a0 = load_afrag(zr, kb + 64, g);
        #pragma unroll
        for (int j = 0; j < 4; j++) b0[j] = load_bfrag(ocol[j], kb + 64);
        #pragma unroll
        for (int j = 0; j < 4; j++)
            acc[j] = __builtin_amdgcn_wmma_f32_16x16x32_f16(false, a1, false, b1[j],
                                                            (short)0, acc[j], false, false);
    }
    a1 = load_afrag(zr, DM - 32, g);
    #pragma unroll
    for (int j = 0; j < 4; j++) b1[j] = load_bfrag(ocol[j], DM - 32);
    #pragma unroll
    for (int j = 0; j < 4; j++)
        acc[j] = __builtin_amdgcn_wmma_f32_16x16x32_f16(false, a0, false, b0[j],
                                                        (short)0, acc[j], false, false);
    #pragma unroll
    for (int j = 0; j < 4; j++)
        acc[j] = __builtin_amdgcn_wmma_f32_16x16x32_f16(false, a1, false, b1[j],
                                                        (short)0, acc[j], false, false);

    #pragma unroll
    for (int j = 0; j < 4; j++) {
        const float bv = Ob[d0 + j*16 + nn];
        #pragma unroll
        for (int i = 0; i < 8; i++)
            out[(size_t)(r0 + i + 8*g) * DM + d0 + j*16 + nn] = acc[j][i] + bv;
    }
}

// ---------------------------------------------------------------------------
extern "C" void kernel_launch(void* const* d_in, const int* in_sizes, int n_in,
                              void* d_out, int out_size, void* d_ws, size_t ws_size,
                              hipStream_t stream)
{
    const float* x   = (const float*)d_in[0];
    const float* Qs  = (const float*)d_in[1];
    const float* Qbs = (const float*)d_in[2];
    const float* Ks  = (const float*)d_in[3];
    const float* Kbs = (const float*)d_in[4];
    const float* Vs  = (const float*)d_in[5];
    const float* Vbs = (const float*)d_in[6];
    const float* O   = (const float*)d_in[7];
    const float* Ob  = (const float*)d_in[8];
    float* out = (float*)d_out;

    const size_t MX = (size_t)2 * SEQ * DM;   // 4M halves
    const size_t MW = (size_t)NH * DM * DH;   // 1M halves

    _Float16* p    = (_Float16*)d_ws;
    _Float16* xh   = p;           p += MX;
    _Float16* wq   = p;           p += MW;
    _Float16* wk   = p;           p += MW;
    _Float16* wv   = p;           p += MW;
    _Float16* ot   = p;           p += MW;
    _Float16* qbuf = p;           p += MX;
    _Float16* kbuf = p;           p += MX;
    _Float16* vtb  = p;           p += MX;
    _Float16* zbuf = p;           p += MX;    // total 24M halves = 48 MB

    // 0) convert x to f16; transpose+convert weights
    prep_kernel<<<32768, 256, 0, stream>>>(x, Qs, Ks, Vs, O, xh, wq, wk, wv, ot);
    // 1) QKV projections: 3 * 2 * 128 * 16 = 12288 waves, 4 waves/block
    qkv_proj_kernel<<<3072, 128, 0, stream>>>(xh, wq, wk, wv, Qbs, Kbs, Vbs,
                                              qbuf, kbuf, vtb);
    // 2) causal flash attention: 2 * 16 * 128 = 4096 waves, 1 wave/block
    attn_kernel<<<4096, 32, 0, stream>>>(qbuf, kbuf, vtb, zbuf);
    // 3) output projection: 256 * 16 = 4096 waves, 4 waves/block
    out_proj_kernel<<<1024, 128, 0, stream>>>(zbuf, ot, Ob, out);
}
